// EncoderBlock_60533269070016
// MI455X (gfx1250) — matrix operand order repaired and checked by reading, not compiled, not run on hardware
//
#include <hip/hip_runtime.h>
#include <hip/hip_bf16.h>

// ---------------------------------------------------------------------------
// MI455X (gfx1250) encoder block: LN1 -> QV proj -> flash attn (k==q quirk,
// scale = D^-1/2) -> Wo+resid -> LN2 -> FC1+GELU -> FC2+resid.
// bf16 WMMA (v_wmma_f32_16x16x32_bf16), f32 accumulation, double-buffered LDS.
// ---------------------------------------------------------------------------

typedef __bf16 bf16;
typedef __attribute__((ext_vector_type(16))) __bf16 v16bf;
typedef __attribute__((ext_vector_type(8)))  __bf16 v8bf;
typedef __attribute__((ext_vector_type(4)))  __bf16 v4bf;
typedef __attribute__((ext_vector_type(8)))  float  v8f;

#define BB 8
#define SS 1024
#define DD 768
#define HH 12
#define EE 64
#define FF (4 * DD)     // 3072
#define MROWS (BB * SS) // 8192

__device__ __forceinline__ float gelu_exact(float x) {
    return 0.5f * x * (1.0f + erff(x * 0.70710678118654752f));
}

// --- WMMA fragment helpers (layouts per CDNA5 ISA 7.12.2, wave32) ----------

// A matrix 16x32 bf16: lane l -> row m=l&15, half=l>>4.
// elements [0..7]  = K = half*8 .. half*8+7      (VGPR0..3)
// elements [8..15] = K = 16+half*8 .. 16+half*8+7 (VGPR4..7)
__device__ __forceinline__ v16bf load_frag_a(const bf16* base, int ld) {
    const int l = threadIdx.x & 31;
    const int m = l & 15, half = l >> 4;
    const bf16* p = base + (size_t)m * ld + half * 8;
    union { v16bf v; v8bf h[2]; } u;
    u.h[0] = *(const v8bf*)p;
    u.h[1] = *(const v8bf*)(p + 16);
    return u.v;
}

// B matrix 32x16 bf16, sourced from B^T stored [N][K] row-major:
// lane l -> col n=l&15, half=l>>4; holds K = half*8..half*8+7, 16+half*8..16+half*8+7.
__device__ __forceinline__ v16bf load_frag_b(const bf16* baseT, int ld) {
    const int l = threadIdx.x & 31;
    const int n = l & 15, half = l >> 4;
    const bf16* p = baseT + (size_t)n * ld + half * 8;
    union { v16bf v; v8bf h[2]; } u;
    u.h[0] = *(const v8bf*)p;
    u.h[1] = *(const v8bf*)(p + 16);
    return u.v;
}

__device__ __forceinline__ v8f wmma_bf16(v16bf a, v16bf b, v8f c) {
    return __builtin_amdgcn_wmma_f32_16x16x32_bf16(
        false, a, false, b, (short)0, c, false, false);
}

// --- LayerNorm (f32 in, bf16 out), one block per row -----------------------
__global__ __launch_bounds__(256)
void layernorm_to_bf16(const float* __restrict__ x, const float* __restrict__ g,
                       const float* __restrict__ beta, bf16* __restrict__ out) {
    __shared__ float red[256];
    const int row = blockIdx.x, tid = threadIdx.x;
    const float* xr = x + (size_t)row * DD;
    float s = 0.f;
    #pragma unroll
    for (int i = 0; i < 3; ++i) s += xr[tid + i * 256];
    red[tid] = s; __syncthreads();
    #pragma unroll
    for (int st = 128; st > 0; st >>= 1) {
        if (tid < st) red[tid] += red[tid + st];
        __syncthreads();
    }
    const float mu = red[0] * (1.0f / DD);
    __syncthreads();
    float s2 = 0.f;
    #pragma unroll
    for (int i = 0; i < 3; ++i) { float d = xr[tid + i * 256] - mu; s2 += d * d; }
    red[tid] = s2; __syncthreads();
    #pragma unroll
    for (int st = 128; st > 0; st >>= 1) {
        if (tid < st) red[tid] += red[tid + st];
        __syncthreads();
    }
    const float rstd = rsqrtf(red[0] * (1.0f / DD) + 1e-5f);
    #pragma unroll
    for (int i = 0; i < 3; ++i) {
        const int c = tid + i * 256;
        out[(size_t)row * DD + c] = (bf16)((xr[c] - mu) * rstd * g[c] + beta[c]);
    }
}

// --- Weight repack: (K,N) f32 row-major -> (N,K) bf16 ----------------------
__global__ __launch_bounds__(256)
void transpose_to_bf16(const float* __restrict__ in, bf16* __restrict__ out,
                       int K, int N) {
    size_t idx = (size_t)blockIdx.x * 256 + threadIdx.x;
    if (idx >= (size_t)K * N) return;
    int k = (int)(idx / N), n = (int)(idx % N);
    out[(size_t)n * K + k] = (bf16)in[idx];
}

// --- Head-weight repack: (H,D,E) f32 -> (H*E, D) bf16 ----------------------
__global__ __launch_bounds__(256)
void repack_head_w(const float* __restrict__ in, bf16* __restrict__ out) {
    size_t idx = (size_t)blockIdx.x * 256 + threadIdx.x;
    if (idx >= (size_t)HH * DD * EE) return;
    int h = (int)(idx / (DD * EE));
    int r = (int)(idx % (DD * EE));
    int d = r / EE, e = r % EE;
    out[(size_t)(h * EE + e) * DD + d] = (bf16)in[idx];
}

// --- Tiled bf16 WMMA GEMM: C[M,N] = A[M,K] @ Bt[N,K]^T + bias (+epilogue) --
// 128x128 block tile, BK=32, double-buffered LDS, 8 waves in 4(M)x2(N),
// each wave 32x64 (2x4 fragments, 8 WMMA per K-step).
// EPI 0: +bias, write bf16.  EPI 1: +bias +resid, write f32.  EPI 2: GELU.
template <int EPI>
__global__ __launch_bounds__(256)
void gemm_bf16_wmma(const bf16* __restrict__ A, const bf16* __restrict__ Bt,
                    const float* __restrict__ bias,
                    const float* __restrict__ resid,
                    void* __restrict__ outv, int M, int N, int K) {
    __shared__ __align__(16) bf16 sA[2][128 * 32];
    __shared__ __align__(16) bf16 sB[2][128 * 32];
    const int tid = threadIdx.x;
    const int wave = tid >> 5;
    const int wm = wave & 3, wn = wave >> 2;   // 4x2 waves -> 128x128 tile
    const int m0 = blockIdx.y * 128, n0 = blockIdx.x * 128;
    // staging map: each thread moves 16 bf16 of A and of B per K-step
    const int sr = tid >> 1, sc = (tid & 1) * 16;
    const bf16* gA = A + (size_t)(m0 + sr) * K + sc;
    const bf16* gB = Bt + (size_t)(n0 + sr) * K + sc;

    v8f c[2][4] = {};
    v8bf ar0, ar1, br0, br1;

    // prolog: tile 0 -> regs -> LDS buf0
    ar0 = *(const v8bf*)(gA);     ar1 = *(const v8bf*)(gA + 8);
    br0 = *(const v8bf*)(gB);     br1 = *(const v8bf*)(gB + 8);
    *(v8bf*)(sA[0] + sr * 32 + sc) = ar0;
    *(v8bf*)(sA[0] + sr * 32 + sc + 8) = ar1;
    *(v8bf*)(sB[0] + sr * 32 + sc) = br0;
    *(v8bf*)(sB[0] + sr * 32 + sc + 8) = br1;
    __syncthreads();

    const int KT = K >> 5;
    for (int kt = 0; kt < KT; ++kt) {
        const int cur = kt & 1, nxt = cur ^ 1;
        if (kt + 1 < KT) {   // prefetch next tile into registers
            const int k0 = (kt + 1) << 5;
            ar0 = *(const v8bf*)(gA + k0);
            ar1 = *(const v8bf*)(gA + k0 + 8);
            br0 = *(const v8bf*)(gB + k0);
            br1 = *(const v8bf*)(gB + k0 + 8);
        }
        // load all fragments, then issue the WMMA group
        const bf16* sAc = sA[cur];
        const bf16* sBc = sB[cur];
        const v16bf a0 = load_frag_a(sAc + (wm * 32) * 32, 32);
        const v16bf a1 = load_frag_a(sAc + (wm * 32 + 16) * 32, 32);
        v16bf b[4];
        #pragma unroll
        for (int j = 0; j < 4; ++j)
            b[j] = load_frag_b(sBc + (wn * 64 + j * 16) * 32, 32);
        #pragma unroll
        for (int j = 0; j < 4; ++j) {
            c[0][j] = wmma_bf16(a0, b[j], c[0][j]);
            c[1][j] = wmma_bf16(a1, b[j], c[1][j]);
        }
        if (kt + 1 < KT) {
            *(v8bf*)(sA[nxt] + sr * 32 + sc) = ar0;
            *(v8bf*)(sA[nxt] + sr * 32 + sc + 8) = ar1;
            *(v8bf*)(sB[nxt] + sr * 32 + sc) = br0;
            *(v8bf*)(sB[nxt] + sr * 32 + sc + 8) = br1;
        }
        __syncthreads();
    }

    // epilogue: C layout -> lane holds col n=l&15, rows half*8 + 0..7
    const int l = tid & 31;
    const int n = l & 15, half = l >> 4;
    #pragma unroll
    for (int i = 0; i < 2; ++i)
        #pragma unroll
        for (int j = 0; j < 4; ++j) {
            const int col = n0 + wn * 64 + j * 16 + n;
            const float bv = bias[col];
            #pragma unroll
            for (int r = 0; r < 8; ++r) {
                const int row = m0 + wm * 32 + i * 16 + half * 8 + r;
                float v = c[i][j][r] + bv;
                if (EPI == 2) v = gelu_exact(v);
                if (EPI == 1) {
                    v += resid[(size_t)row * N + col];
                    ((float*)outv)[(size_t)row * N + col] = v;
                } else {
                    ((bf16*)outv)[(size_t)row * N + col] = (bf16)v;
                }
            }
        }
}

// --- Fused flash attention (k == q quirk; scale = D^-1/2) ------------------
// block = 8 waves; each wave owns 16 queries x E=64; keys streamed 64 at a
// time with double-buffered K / V^T tiles. One barrier and one 4-stage
// max/sum butterfly per row per 64 keys. P staged per-wave in LDS with the
// 64 keys interleaved (perm 4*(key&15)+(key>>4)) so all four probabilities
// per lane pack into one 8-byte LDS store; V^T columns staged with the
// matching permutation (softmax @ V is key-permutation invariant).
__global__ __launch_bounds__(256)
void flash_attn_bf16(const bf16* __restrict__ q, const bf16* __restrict__ v,
                     bf16* __restrict__ o_out, float scale) {
    __shared__ __align__(16) bf16 sK[2][64 * 64];   // [key][E]
    __shared__ __align__(16) bf16 sVT[2][64 * 64];  // [E][perm(key)]
    __shared__ __align__(16) bf16 sP[8][16 * 64];   // per-wave P staging
    const int tid = threadIdx.x;
    const int wave = tid >> 5, l = tid & 31;
    const int n = l & 15, half = l >> 4;
    const int b = blockIdx.z, h = blockIdx.y;
    const int qs = blockIdx.x * 128 + wave * 16;

    const bf16* qbase = q + (size_t)(b * SS + qs) * DD + h * EE;
    const v16bf qa0 = load_frag_a(qbase, DD);        // E = 0..31
    const v16bf qa1 = load_frag_a(qbase + 32, DD);   // E = 32..63

    // staging map (16 bf16 per thread for each of K and V; 64x64 tile)
    const int kr = tid >> 2, kc = (tid & 3) * 16;
    const int vperm = 4 * (kr & 15) + (kr >> 4);     // permuted key column
    const bf16* gK = q + (size_t)(b * SS + kr) * DD + h * EE + kc;
    const bf16* gV = v + (size_t)(b * SS + kr) * DD + h * EE + kc;

    v8f o[4] = {};
    float mrow[8], lrow[8];
    #pragma unroll
    for (int i = 0; i < 8; ++i) { mrow[i] = -3.0e38f; lrow[i] = 0.f; }

    // prolog: tile 0 -> regs -> LDS buf0
    v8bf kg0 = *(const v8bf*)gK, kg1 = *(const v8bf*)(gK + 8);
    v8bf vg0 = *(const v8bf*)gV, vg1 = *(const v8bf*)(gV + 8);
    *(v8bf*)(sK[0] + kr * 64 + kc) = kg0;
    *(v8bf*)(sK[0] + kr * 64 + kc + 8) = kg1;
    #pragma unroll
    for (int j = 0; j < 8; ++j) {
        sVT[0][(kc + j) * 64 + vperm] = vg0[j];
        sVT[0][(kc + 8 + j) * 64 + vperm] = vg1[j];
    }
    __syncthreads();

    const int NT = SS / 64;
    for (int kb = 0; kb < NT; ++kb) {
        const int cur = kb & 1, nxt = cur ^ 1;
        if (kb + 1 < NT) {   // prefetch next K/V tile into registers
            const size_t off = (size_t)(kb + 1) * 64 * DD;
            kg0 = *(const v8bf*)(gK + off);
            kg1 = *(const v8bf*)(gK + off + 8);
            vg0 = *(const v8bf*)(gV + off);
            vg1 = *(const v8bf*)(gV + off + 8);
        }

        // scores: 16 queries x 64 keys (keys ARE q), 4 chunks of 16
        v8f s[4] = {};
        {
            const bf16* kt = sK[cur];
            #pragma unroll
            for (int cch = 0; cch < 4; ++cch) {
                const v16bf kb0 = load_frag_b(kt + (cch * 16) * 64, 64);
                const v16bf kb1 = load_frag_b(kt + (cch * 16) * 64 + 32, 64);
                s[cch] = wmma_bf16(qa0, kb0, s[cch]);
                s[cch] = wmma_bf16(qa1, kb1, s[cch]);
            }
        }

        // online softmax over 64 new columns; packed P -> LDS
        bf16* pw = sP[wave];
        #pragma unroll
        for (int i = 0; i < 8; ++i) {
            const float a0 = s[0][i] * scale, a1 = s[1][i] * scale;
            const float a2 = s[2][i] * scale, a3 = s[3][i] * scale;
            float rmax = fmaxf(fmaxf(a0, a1), fmaxf(a2, a3));
            rmax = fmaxf(rmax, __shfl_xor(rmax, 1));
            rmax = fmaxf(rmax, __shfl_xor(rmax, 2));
            rmax = fmaxf(rmax, __shfl_xor(rmax, 4));
            rmax = fmaxf(rmax, __shfl_xor(rmax, 8));
            const float mnew = fmaxf(mrow[i], rmax);
            const float alpha = __expf(mrow[i] - mnew);
            const float p0 = __expf(a0 - mnew), p1 = __expf(a1 - mnew);
            const float p2 = __expf(a2 - mnew), p3 = __expf(a3 - mnew);
            float rs = (p0 + p1) + (p2 + p3);
            rs += __shfl_xor(rs, 1); rs += __shfl_xor(rs, 2);
            rs += __shfl_xor(rs, 4); rs += __shfl_xor(rs, 8);
            lrow[i] = lrow[i] * alpha + rs;
            mrow[i] = mnew;
            o[0][i] *= alpha; o[1][i] *= alpha;
            o[2][i] *= alpha; o[3][i] *= alpha;
            v4bf pk;
            pk[0] = (bf16)p0; pk[1] = (bf16)p1;
            pk[2] = (bf16)p2; pk[3] = (bf16)p3;   // cols 4n .. 4n+3
            *(v4bf*)(pw + (half * 8 + i) * 64 + 4 * n) = pk;
        }

        // O += P(16x64) @ V(64x64): 2 K-steps x 4 N-chunks (key order
        // permuted identically in P and V^T)
        const v16bf pa0 = load_frag_a(pw, 64);
        const v16bf pa1 = load_frag_a(pw + 32, 64);
        {
            const bf16* vt = sVT[cur];
            #pragma unroll
            for (int cch = 0; cch < 4; ++cch) {
                o[cch] = wmma_bf16(pa0, load_frag_b(vt + (cch * 16) * 64, 64), o[cch]);
                o[cch] = wmma_bf16(pa1, load_frag_b(vt + (cch * 16) * 64 + 32, 64), o[cch]);
            }
        }

        if (kb + 1 < NT) {   // regs -> LDS buf nxt
            *(v8bf*)(sK[nxt] + kr * 64 + kc) = kg0;
            *(v8bf*)(sK[nxt] + kr * 64 + kc + 8) = kg1;
            #pragma unroll
            for (int j = 0; j < 8; ++j) {
                sVT[nxt][(kc + j) * 64 + vperm] = vg0[j];
                sVT[nxt][(kc + 8 + j) * 64 + vperm] = vg1[j];
            }
        }
        __syncthreads();
    }

    // normalize and write concat-head layout [b*S+s][h*E+e] as bf16
    #pragma unroll
    for (int i = 0; i < 8; ++i) {
        const float inv = 1.0f / lrow[i];
        const size_t row = (size_t)(b * SS + qs + half * 8 + i) * DD + h * EE + n;
        o_out[row + 0]  = (bf16)(o[0][i] * inv);
        o_out[row + 16] = (bf16)(o[1][i] * inv);
        o_out[row + 32] = (bf16)(o[2][i] * inv);
        o_out[row + 48] = (bf16)(o[3][i] * inv);
    }
}

// ---------------------------------------------------------------------------
extern "C" void kernel_launch(void* const* d_in, const int* in_sizes, int n_in,
                              void* d_out, int out_size, void* d_ws, size_t ws_size,
                              hipStream_t stream) {
    const float* x     = (const float*)d_in[0];
    const float* ln1_g = (const float*)d_in[1];
    const float* ln1_b = (const float*)d_in[2];
    const float* Wq    = (const float*)d_in[3];
    const float* bq    = (const float*)d_in[4];
    const float* Wv    = (const float*)d_in[5];
    const float* bv    = (const float*)d_in[6];
    const float* Wo    = (const float*)d_in[7];
    const float* bo    = (const float*)d_in[8];
    const float* ln2_g = (const float*)d_in[9];
    const float* ln2_b = (const float*)d_in[10];
    const float* W1    = (const float*)d_in[11];
    const float* b1    = (const float*)d_in[12];
    const float* W2    = (const float*)d_in[13];
    const float* b2    = (const float*)d_in[14];
    float* out = (float*)d_out;

    char* w = (char*)d_ws;
    auto carve = [&](size_t bytes) {
        char* p = w;
        w += (bytes + 255) & ~(size_t)255;
        return p;
    };
    bf16* hb    = (bf16*)carve((size_t)MROWS * DD * 2);
    bf16* qb    = (bf16*)carve((size_t)MROWS * DD * 2);
    bf16* vb    = (bf16*)carve((size_t)MROWS * DD * 2);
    bf16* ab    = (bf16*)carve((size_t)MROWS * DD * 2);   // attn concat out
    float* x2   = (float*)carve((size_t)MROWS * DD * 4);  // post-Wo residual
    bf16* h2b   = (bf16*)carve((size_t)MROWS * DD * 2);
    bf16* mlpb  = (bf16*)carve((size_t)MROWS * FF * 2);   // gelu(fc1)
    bf16* Wq_t  = (bf16*)carve((size_t)DD * DD * 2);
    bf16* Wv_t  = (bf16*)carve((size_t)DD * DD * 2);
    bf16* Wo_t  = (bf16*)carve((size_t)DD * DD * 2);
    bf16* W1_t  = (bf16*)carve((size_t)DD * FF * 2);
    bf16* W2_t  = (bf16*)carve((size_t)FF * DD * 2);

    // 1) LN1
    layernorm_to_bf16<<<MROWS, 256, 0, stream>>>(x, ln1_g, ln1_b, hb);

    // 2) weight repacks (bf16, [N][K])
    {
        const int nqv = HH * DD * EE;
        repack_head_w<<<(nqv + 255) / 256, 256, 0, stream>>>(Wq, Wq_t);
        repack_head_w<<<(nqv + 255) / 256, 256, 0, stream>>>(Wv, Wv_t);
        transpose_to_bf16<<<(DD * DD + 255) / 256, 256, 0, stream>>>(Wo, Wo_t, DD, DD);
        transpose_to_bf16<<<(DD * FF + 255) / 256, 256, 0, stream>>>(W1, W1_t, DD, FF);
        transpose_to_bf16<<<(FF * DD + 255) / 256, 256, 0, stream>>>(W2, W2_t, FF, DD);
    }

    // 3) Q and V projections (bias, bf16 out)
    {
        dim3 grid(DD / 128, MROWS / 128);
        gemm_bf16_wmma<0><<<grid, 256, 0, stream>>>(hb, Wq_t, bq, nullptr, qb,
                                                    MROWS, DD, DD);
        gemm_bf16_wmma<0><<<grid, 256, 0, stream>>>(hb, Wv_t, bv, nullptr, vb,
                                                    MROWS, DD, DD);
    }

    // 4) fused flash attention (keys == queries per reference quirk)
    {
        const float scale = 0.036084391824351615f; // 1/sqrt(768) — full d_model
        dim3 grid(SS / 128, HH, BB);
        flash_attn_bf16<<<grid, 256, 0, stream>>>(qb, vb, ab, scale);
    }

    // 5) output projection + residual(x) -> x2 (f32)
    {
        dim3 grid(DD / 128, MROWS / 128);
        gemm_bf16_wmma<1><<<grid, 256, 0, stream>>>(ab, Wo_t, bo, x, x2,
                                                    MROWS, DD, DD);
    }

    // 6) LN2
    layernorm_to_bf16<<<MROWS, 256, 0, stream>>>(x2, ln2_g, ln2_b, h2b);

    // 7) FC1 + exact GELU (bf16 out)
    {
        dim3 grid(FF / 128, MROWS / 128);
        gemm_bf16_wmma<2><<<grid, 256, 0, stream>>>(h2b, W1_t, b1, nullptr, mlpb,
                                                    MROWS, FF, DD);
    }

    // 8) FC2 + residual(x2) -> d_out (f32)
    {
        dim3 grid(DD / 128, MROWS / 128);
        gemm_bf16_wmma<1><<<grid, 256, 0, stream>>>(mlpb, W2_t, b2, x2, out,
                                                    MROWS, DD, FF);
    }
    (void)in_sizes; (void)n_in; (void)out_size; (void)ws_size;
}
